// AttentionBlock_17085379904404
// MI455X (gfx1250) — matrix operand order repaired and checked by reading, not compiled, not run on hardware
//
#include <hip/hip_runtime.h>
#include <hip/hip_bf16.h>

typedef __attribute__((ext_vector_type(16))) _Float16 v16h;
typedef __attribute__((ext_vector_type(8)))  float    v8f;

#define WMMA_F32_F16(a, b, c) \
  __builtin_amdgcn_wmma_f32_16x16x32_f16(false, (a), false, (b), (short)0, (c), false, false)

// ---------------- CDNA5 async LDS staging (CDNA5 ISA, GV mode) ----------------
// Copies 16 bytes per lane: LDS[lds_off] = MEM[gaddr]; tracked with ASYNCcnt.
__device__ __forceinline__ void async_ld_b128(unsigned lds_off, const void* gptr) {
  asm volatile("global_load_async_to_lds_b128 %0, %1, off"
               :: "v"(lds_off), "v"((unsigned long long)gptr) : "memory");
}
// Wait until ASYNCcnt <= n (async loads complete in order -> partial waits pipeline safely).
#define WAIT_ASYNC(n) asm volatile("s_wait_asynccnt " #n ::: "memory")

// ---------------- WMMA fragment helpers (CDNA5 16x16x32 f16 layouts) ----------------
// A matrix 16x32 (MxK), tile stored row-major [M][K] with stride ld (halves).
__device__ __forceinline__ v16h load_A_frag(const _Float16* a, int ld) {
  const int lane = threadIdx.x & 31;
  const int row  = lane & 15;
  const int kb   = (lane >> 4) << 3;          // 0 or 8
  const _Float16* p = a + row * ld + kb;
  v16h r;
  *(uint4*)&r         = *(const uint4*)p;        // K = kb .. kb+7
  *(((uint4*)&r) + 1) = *(const uint4*)(p + 16); // K = kb+16 .. kb+23
  return r;
}

// B matrix 32x16 (KxN), tile stored N-major [N][K] with stride ld (halves):
// 2x16B per lane. col = lane&15 ; lanes<16 hold K 0-7,16-23 ; lanes>=16 K 8-15,24-31.
__device__ __forceinline__ v16h load_B_frag_nk(const _Float16* b, int ld) {
  const int lane = threadIdx.x & 31;
  const int col  = lane & 15;
  const int kb   = (lane >> 4) << 3;          // 0 or 8
  const _Float16* p = b + col * ld + kb;
  v16h r;
  *(uint4*)&r         = *(const uint4*)p;
  *(((uint4*)&r) + 1) = *(const uint4*)(p + 16);
  return r;
}

// ---------------- Constants ----------------
#define B_   32
#define C_   512
#define N_   1024          // H*W
#define OQKV 1536          // 3*C
#define OQK  1024          // q,k channels in token-major buffer

// ---------------- 1) GroupNorm -> h (f16, token-major [B][N][C]) ----------------
__global__ __launch_bounds__(256) void groupnorm_kernel(
    const float* __restrict__ x, const float* __restrict__ gw,
    const float* __restrict__ gb, _Float16* __restrict__ h) {
  const int bg = blockIdx.x;             // b*32 + g
  const int b = bg >> 5, g = bg & 31;
  const float* xp = x + (size_t)b * C_ * N_ + (size_t)g * 16 * N_;
  __shared__ float r1[256], r2[256];
  float s = 0.f, s2 = 0.f;
  for (int i = threadIdx.x; i < 16 * N_; i += 256) {
    float v = xp[i]; s += v; s2 += v * v;
  }
  r1[threadIdx.x] = s; r2[threadIdx.x] = s2;
  __syncthreads();
  for (int st = 128; st; st >>= 1) {
    if (threadIdx.x < st) { r1[threadIdx.x] += r1[threadIdx.x + st];
                            r2[threadIdx.x] += r2[threadIdx.x + st]; }
    __syncthreads();
  }
  const float mean = r1[0] * (1.f / 16384.f);
  const float var  = r2[0] * (1.f / 16384.f) - mean * mean;
  const float rstd = rsqrtf(var + 1e-5f);
  for (int i = threadIdx.x; i < 16 * N_; i += 256) {
    const int cc = i >> 10, n = i & (N_ - 1);
    const int c = g * 16 + cc;
    const float v = (xp[i] - mean) * rstd * gw[c] + gb[c];
    h[(size_t)b * N_ * C_ + (size_t)n * C_ + c] = (_Float16)v;
  }
}

// ---------------- 2) f32 -> f16 convert (weights keep native [o][c] layout) --------
__global__ __launch_bounds__(256) void convert_f16(
    const float* __restrict__ src, _Float16* __restrict__ dst, int n) {
  const int i = blockIdx.x * 256 + threadIdx.x;
  if (i < n) dst[i] = (_Float16)src[i];
}

// ======== shared double-buffered GEMM core: 64x64 tile, 4 waves, K pipelined ========
// LDS layout (dynamic, 20480 B): A0 @0, A1 @5120, B0 @10240, B1 @15360 (64x40 halves each)
__device__ __forceinline__ void gemm_issue_tile(
    const _Float16* hA, const _Float16* wB, int k0, unsigned buf, int tid) {
  const unsigned bo = buf * 5120u;
  for (int idx = tid; idx < 256; idx += 128) {           // A tile 64 x 32
    const int r = idx >> 2, cc = idx & 3;
    async_ld_b128(bo + r * 80 + cc * 16, &hA[(size_t)r * C_ + k0 + cc * 8]);
  }
  for (int idx = tid; idx < 256; idx += 128) {           // B tile 64(o) x 32(c), N-major
    const int r = idx >> 2, cc = idx & 3;
    async_ld_b128(10240u + bo + r * 80 + cc * 16, &wB[(size_t)r * C_ + k0 + cc * 8]);
  }
}

__device__ __forceinline__ void gemm_core(
    char* smem, const _Float16* hA, const _Float16* wB, int tid, int wave, v8f acc[4]) {
  gemm_issue_tile(hA, wB, 0, 0, tid);
  for (int step = 0; step < 16; ++step) {
    const unsigned buf = step & 1;
    if (step + 1 < 16) {
      gemm_issue_tile(hA, wB, (step + 1) * 32, (step + 1) & 1, tid);
      WAIT_ASYNC(0x4);                 // prefetch (4 asyncs/wave) still in flight
    } else {
      WAIT_ASYNC(0x0);
    }
    __syncthreads();                   // current tile visible to all waves
    const _Float16* As = (const _Float16*)(smem + buf * 5120);
    const _Float16* Bs = (const _Float16*)(smem + 10240 + buf * 5120);
    const v16h a = load_A_frag(As + wave * 16 * 40, 40);
#pragma unroll
    for (int f = 0; f < 4; ++f) {
      const v16h bb = load_B_frag_nk(Bs + f * 16 * 40, 40);
      acc[f] = WMMA_F32_F16(a, bb, acc[f]);
    }
    __syncthreads();                   // done reading buf before it is refilled
  }
}

// ---------------- 3) QKV GEMM ----------------
// qk[b][n][o] (o<1024: q,k token-major) and Vt[b][c][m] (v channel-major, packed stores).
__global__ __launch_bounds__(128) void qkv_gemm(
    const _Float16* __restrict__ h, const _Float16* __restrict__ w,
    const float* __restrict__ bias, _Float16* __restrict__ qk,
    _Float16* __restrict__ vt) {
  extern __shared__ char smem[];
  const int mt = blockIdx.x, nt = blockIdx.y, b = blockIdx.z;
  const int tid = threadIdx.x, wave = tid >> 5, lane = tid & 31;
  const _Float16* hA = h + (size_t)b * N_ * C_ + (size_t)mt * 64 * C_;
  const _Float16* wB = w + (size_t)nt * 64 * C_;      // rows o, ld = C_
  v8f acc[4] = {};
  gemm_core(smem, hA, wB, tid, wave, acc);
  const int col = lane & 15, mb = (lane >> 4) * 8;
#pragma unroll
  for (int f = 0; f < 4; ++f) {
    const int o = nt * 64 + f * 16 + col;
    const float bv = bias[o];
    if (nt * 64 + f * 16 >= 1024) {
      // V third: channel-major, tokens contiguous -> packed 16B store
      union { _Float16 t[8]; uint4 u; } pk;
#pragma unroll
      for (int i = 0; i < 8; ++i) pk.t[i] = (_Float16)(acc[f][i] + bv);
      const int token0 = mt * 64 + wave * 16 + mb;
      *(uint4*)&vt[(size_t)b * C_ * N_ + (size_t)(o - 1024) * N_ + token0] = pk.u;
    } else {
#pragma unroll
      for (int i = 0; i < 8; ++i) {
        const int token = mt * 64 + wave * 16 + mb + i;
        qk[(size_t)b * N_ * OQK + (size_t)token * OQK + o] = (_Float16)(acc[f][i] + bv);
      }
    }
  }
}

// ---------------- 4) fused attention: scores + softmax + P*V^T ----------------
// One workgroup (8 waves) per 16-query tile; dynamic LDS partitions (bytes):
#define LDQ 520            // halves
#define LDK 520            // halves (K block 128 x 512)
#define LDV 136            // halves (V block 512 x 128, channel-major)
#define LDSS 1032          // f32
#define LDP 1040           // halves
#define SM_Q 0             // 16*520*2          = 16640
#define SM_K 16640         // max(128*520*2, 512*136*2) = 139264
#define SM_S 155904        // 16*1032*4         = 66048
#define SM_P 221952        // 16*1040*2         = 33280
#define SM_TOTAL 255232

__global__ __launch_bounds__(256) void attn_kernel(
    const _Float16* __restrict__ qk, const _Float16* __restrict__ vt,
    _Float16* __restrict__ houtT) {
  extern __shared__ char smem[];
  _Float16* Qs = (_Float16*)(smem + SM_Q);
  _Float16* Ks = (_Float16*)(smem + SM_K);
  _Float16* Vs = (_Float16*)(smem + SM_K);
  float*    Ss = (float*)(smem + SM_S);
  _Float16* Ps = (_Float16*)(smem + SM_P);
  const int qt = blockIdx.x, b = blockIdx.y;
  const int tid = threadIdx.x, wave = tid >> 5, lane = tid & 31;
  const _Float16* qkb = qk + (size_t)b * N_ * OQK;
  const _Float16* vtb = vt + (size_t)b * C_ * N_;
  const float scale = 0.044194173824159216f;   // 512^-0.5

  // Q tile: 16 x 512 (async -> LDS)
  for (int idx = tid; idx < 1024; idx += 256) {
    const int r = idx >> 6, cc = idx & 63;
    async_ld_b128(SM_Q + r * 1040 + cc * 16,
                  &qkb[(size_t)(qt * 16 + r) * OQK + cc * 8]);
  }

  // ---- pass 1: S = scale * Q K^T, 8 key-blocks of 128 ----
  for (int kb = 0; kb < 8; ++kb) {
    __syncthreads();
    for (int idx = tid; idx < 8192; idx += 256) {      // K block 128 x 512, token-major
      const int r = idx >> 6, cc = idx & 63;
      async_ld_b128(SM_K + r * 1040 + cc * 16,
                    &qkb[(size_t)(kb * 128 + r) * OQK + C_ + cc * 8]);
    }
    WAIT_ASYNC(0x0);
    __syncthreads();
    v8f s = {};
#pragma unroll
    for (int ks = 0; ks < 16; ++ks) {
      const v16h a  = load_A_frag(Qs + ks * 32, LDQ);
      const v16h bb = load_B_frag_nk(Ks + (wave * 16) * LDK + ks * 32, LDK);
      s = WMMA_F32_F16(a, bb, s);
    }
    const int col = lane & 15, mb = (lane >> 4) * 8;
#pragma unroll
    for (int i = 0; i < 8; ++i)
      Ss[(mb + i) * LDSS + kb * 128 + wave * 16 + col] = s[i] * scale;
  }
  __syncthreads();

  // ---- pass 2: exact softmax over each of the 16 rows (16 lanes per row) ----
  {
    const int r = tid >> 4, i = tid & 15;
    float m = -1e30f;
    for (int c = i; c < N_; c += 16) m = fmaxf(m, Ss[r * LDSS + c]);
    for (int off = 8; off; off >>= 1) m = fmaxf(m, __shfl_xor(m, off, 32));
    float sum = 0.f;
    for (int c = i; c < N_; c += 16) sum += __expf(Ss[r * LDSS + c] - m);
    for (int off = 8; off; off >>= 1) sum += __shfl_xor(sum, off, 32);
    const float inv = 1.f / sum;
    for (int c = i; c < N_; c += 16)
      Ps[r * LDP + c] = (_Float16)(__expf(Ss[r * LDSS + c] - m) * inv);
  }

  // ---- pass 3: O = P * V^T ; V staged channel-major so B frags are contiguous ----
  v8f acc[4] = {};
  for (int vb = 0; vb < 8; ++vb) {
    __syncthreads();
    for (int idx = tid; idx < 8192; idx += 256) {      // V block 512(c) x 128(m)
      const int r = idx >> 4, cc = idx & 15;
      async_ld_b128(SM_K + r * 272 + cc * 16,
                    &vtb[(size_t)r * N_ + vb * 128 + cc * 8]);
    }
    WAIT_ASYNC(0x0);
    __syncthreads();
#pragma unroll
    for (int ks = 0; ks < 4; ++ks) {
      const v16h a = load_A_frag(Ps + vb * 128 + ks * 32, LDP);
#pragma unroll
      for (int f = 0; f < 4; ++f) {
        const v16h bb = load_B_frag_nk(Vs + (wave * 64 + f * 16) * LDV + ks * 32, LDV);
        acc[f] = WMMA_F32_F16(a, bb, acc[f]);
      }
    }
  }
  const int col = lane & 15, mb = (lane >> 4) * 8;
  _Float16* op = houtT + (size_t)b * N_ * C_ + (size_t)(qt * 16) * C_;
#pragma unroll
  for (int f = 0; f < 4; ++f)
#pragma unroll
    for (int i = 0; i < 8; ++i)
      op[(size_t)(mb + i) * C_ + wave * 64 + f * 16 + col] = (_Float16)acc[f][i];
}

// ---------------- 5) proj GEMM + residual: out[b][o][n] = x + hT.Wproj^T + bias ----
__global__ __launch_bounds__(128) void proj_gemm(
    const _Float16* __restrict__ hT, const _Float16* __restrict__ w,
    const float* __restrict__ bias, const float* __restrict__ x,
    float* __restrict__ out) {
  extern __shared__ char smem[];
  const int mt = blockIdx.x, nt = blockIdx.y, b = blockIdx.z;
  const int tid = threadIdx.x, wave = tid >> 5, lane = tid & 31;
  const _Float16* hA = hT + (size_t)b * N_ * C_ + (size_t)mt * 64 * C_;
  const _Float16* wB = w + (size_t)nt * 64 * C_;      // rows o, ld = C_
  v8f acc[4] = {};
  gemm_core(smem, hA, wB, tid, wave, acc);
  const int col = lane & 15, mb = (lane >> 4) * 8;
#pragma unroll
  for (int f = 0; f < 4; ++f) {
    const int o = nt * 64 + f * 16 + col;
    const float bv = bias[o];
#pragma unroll
    for (int i = 0; i < 8; ++i) {
      const int token = mt * 64 + wave * 16 + mb + i;
      const size_t idx = (size_t)b * C_ * N_ + (size_t)o * N_ + token;
      out[idx] = x[idx] + acc[f][i] + bv;
    }
  }
}

// ---------------- host ----------------
extern "C" void kernel_launch(void* const* d_in, const int* in_sizes, int n_in,
                              void* d_out, int out_size, void* d_ws, size_t ws_size,
                              hipStream_t stream) {
  const float* x      = (const float*)d_in[0];
  const float* norm_w = (const float*)d_in[1];
  const float* norm_b = (const float*)d_in[2];
  const float* qkv_w  = (const float*)d_in[3];
  const float* qkv_b  = (const float*)d_in[4];
  const float* proj_w = (const float*)d_in[5];
  const float* proj_b = (const float*)d_in[6];
  float* out = (float*)d_out;

  char* ws = (char*)d_ws;
  const size_t h_bytes    = (size_t)B_ * N_ * C_ * 2;      // 32 MB
  const size_t qk_bytes   = (size_t)B_ * N_ * OQK * 2;     // 64 MB
  const size_t vt_bytes   = (size_t)B_ * C_ * N_ * 2;      // 32 MB
  const size_t hout_bytes = (size_t)B_ * N_ * C_ * 2;      // 32 MB
  const size_t wq_bytes   = (size_t)OQKV * C_ * 2;         // 1.5 MB
  _Float16* h     = (_Float16*)(ws);
  _Float16* qk    = (_Float16*)(ws + h_bytes);
  _Float16* vt    = (_Float16*)(ws + h_bytes + qk_bytes);
  _Float16* hout  = (_Float16*)(ws + h_bytes + qk_bytes + vt_bytes);
  _Float16* wq16  = (_Float16*)(ws + h_bytes + qk_bytes + vt_bytes + hout_bytes);
  _Float16* wp16  = (_Float16*)(ws + h_bytes + qk_bytes + vt_bytes + hout_bytes + wq_bytes);

  groupnorm_kernel<<<B_ * 32, 256, 0, stream>>>(x, norm_w, norm_b, h);
  convert_f16<<<(OQKV * C_ + 255) / 256, 256, 0, stream>>>(qkv_w, wq16, OQKV * C_);
  convert_f16<<<(C_ * C_ + 255) / 256, 256, 0, stream>>>(proj_w, wp16, C_ * C_);
  qkv_gemm<<<dim3(N_ / 64, OQKV / 64, B_), 128, 20480, stream>>>(h, wq16, qkv_b, qk, vt);
  attn_kernel<<<dim3(N_ / 16, B_), 256, SM_TOTAL, stream>>>(qk, vt, hout);
  proj_gemm<<<dim3(N_ / 64, C_ / 64, B_), 128, 20480, stream>>>(hout, wp16, proj_b, x, out);
}
